// DoubleNet_17437567222411
// MI455X (gfx1250) — compile-verified, matrix operand order checked by reading, not run on hardware
//
#include <hip/hip_runtime.h>
#include <hip/hip_bf16.h>

// ---------------------------------------------------------------------------
// DoubleNet on MI455X (gfx1250): ternary-masked GEMM stack via bf16 WMMA.
// Weights are {-1,0,+1} (exact in bf16); activations bf16, accumulate fp32.
// Top-50% mask thresholds found by 2-pass 16-bit radix select on |score| bits.
// GEMM global->LDS staging uses gfx1250 async-to-LDS (ASYNCcnt) with
// double-buffered LDS tiles so loads overlap the WMMA block.
// ---------------------------------------------------------------------------

typedef __attribute__((ext_vector_type(16))) __bf16 v16bf;
typedef __attribute__((ext_vector_type(8)))  float  v8f;

__device__ __forceinline__ unsigned short f32_to_bf16(float f) {
    unsigned int u = __float_as_uint(f);
    unsigned int r = u + 0x7FFFu + ((u >> 16) & 1u);   // round-to-nearest-even
    return (unsigned short)(r >> 16);
}
__device__ __forceinline__ float bf16_to_f32(unsigned short h) {
    return __uint_as_float(((unsigned int)h) << 16);
}
__device__ __forceinline__ unsigned int abs_bits(float f) {
    return __float_as_uint(f) & 0x7FFFFFFFu;           // monotone key for |f|
}
// Low 32 bits of a generic pointer into LDS = wave-relative LDS byte offset
// (ISA: LDS aperture keeps the offset in addr[31:0]).
__device__ __forceinline__ unsigned int lds_addr32(const void* p) {
    return (unsigned int)(unsigned long long)p;
}

// -------------------------------- histogram / radix select -----------------

__global__ void zero_hist_k(unsigned int* __restrict__ hist) {
    for (unsigned int i = blockIdx.x * blockDim.x + threadIdx.x;
         i < 65536u; i += gridDim.x * blockDim.x)
        hist[i] = 0u;
}

__global__ void hist_hi_k(const float* __restrict__ s, size_t n,
                          unsigned int* __restrict__ hist) {
    size_t stride = (size_t)gridDim.x * blockDim.x;
    for (size_t i = (size_t)blockIdx.x * blockDim.x + threadIdx.x; i < n; i += stride) {
        unsigned int b = abs_bits(s[i]);
        atomicAdd(&hist[b >> 16], 1u);
    }
}

__global__ void select_hi_k(const unsigned int* __restrict__ hist,
                            unsigned int j, unsigned int* __restrict__ sel) {
    if (threadIdx.x == 0 && blockIdx.x == 0) {
        unsigned int cum = 0;
        for (int b = 0; b < 65536; ++b) {
            unsigned int c = hist[b];
            if (cum + c > j) { sel[0] = (unsigned int)b; sel[1] = j - cum; return; }
            cum += c;
        }
        sel[0] = 65535u; sel[1] = 0u;
    }
}

__global__ void hist_lo_k(const float* __restrict__ s, size_t n,
                          unsigned int* __restrict__ hist,
                          const unsigned int* __restrict__ sel) {
    unsigned int bucket = sel[0];
    size_t stride = (size_t)gridDim.x * blockDim.x;
    for (size_t i = (size_t)blockIdx.x * blockDim.x + threadIdx.x; i < n; i += stride) {
        unsigned int b = abs_bits(s[i]);
        if ((b >> 16) == bucket) atomicAdd(&hist[b & 0xFFFFu], 1u);
    }
}

__global__ void select_lo_k(const unsigned int* __restrict__ hist,
                            const unsigned int* __restrict__ sel,
                            unsigned int* __restrict__ tout) {
    if (threadIdx.x == 0 && blockIdx.x == 0) {
        unsigned int rem = sel[1], cum = 0, low = 65535u;
        for (int b = 0; b < 65536; ++b) {
            unsigned int c = hist[b];
            if (cum + c > rem) { low = (unsigned int)b; break; }
            cum += c;
        }
        *tout = (sel[0] << 16) | low;   // 32-bit key threshold (rank-j value)
    }
}

// -------------------------------- weight build ------------------------------

__global__ void build_w_k(const float* __restrict__ s1, const float* __restrict__ s2,
                          const unsigned int* __restrict__ t1,
                          const unsigned int* __restrict__ t2,
                          unsigned short* __restrict__ W, size_t n) {
    unsigned int T1 = *t1, T2 = *t2;
    size_t stride = (size_t)gridDim.x * blockDim.x;
    for (size_t i = (size_t)blockIdx.x * blockDim.x + threadIdx.x; i < n; i += stride) {
        int m = (int)(abs_bits(s1[i]) >= T1) - (int)(abs_bits(s2[i]) >= T2);
        W[i] = (m > 0) ? 0x3F80u : ((m < 0) ? 0xBF80u : 0x0000u);  // bf16 +1/-1/0
    }
}

__global__ void build_wout_k(const float* __restrict__ s1, const float* __restrict__ s2,
                             const unsigned int* __restrict__ t1,
                             const unsigned int* __restrict__ t2,
                             float* __restrict__ W, size_t n) {
    unsigned int T1 = *t1, T2 = *t2;
    size_t stride = (size_t)gridDim.x * blockDim.x;
    for (size_t i = (size_t)blockIdx.x * blockDim.x + threadIdx.x; i < n; i += stride)
        W[i] = (float)((int)(abs_bits(s1[i]) >= T1) - (int)(abs_bits(s2[i]) >= T2));
}

__global__ void cvt_f32_bf16_k(const float* __restrict__ x,
                               unsigned short* __restrict__ o, size_t n) {
    size_t stride = (size_t)gridDim.x * blockDim.x;
    for (size_t i = (size_t)blockIdx.x * blockDim.x + threadIdx.x; i < n; i += stride)
        o[i] = f32_to_bf16(x[i]);
}

// -------------------------------- WMMA GEMM ---------------------------------
// O[M,N] = relu(A[M,K] @ W[N,K]^T + 2*bias), bf16 in / bf16 out, f32 acc.
// Block tile 128x128, BK=32, double-buffered LDS (2 x (8KB A + 8KB W)).
// 8 waves: 4 (M) x 2 (N); wave tile 32x64 = 2x4 frags of
// v_wmma_f32_16x16x32_bf16. Tiles staged via global_load_async_to_lds_b128.

__global__ __launch_bounds__(256, 1)
void gemm_relu_bf16_k(const unsigned short* __restrict__ A,
                      const unsigned short* __restrict__ W,
                      const float* __restrict__ bias,
                      unsigned short* __restrict__ O,
                      int M, int N, int Kd) {
    __shared__ __align__(16) unsigned short As[2][128 * 32];
    __shared__ __align__(16) unsigned short Ws[2][128 * 32];

    const int tid  = threadIdx.x;
    const int lane = tid & 31;
    const int wave = tid >> 5;
    const int wm   = wave & 3;       // M sub-tile (32 rows)
    const int wn   = wave >> 2;      // N sub-tile (64 cols)
    const int m0   = blockIdx.y * 128;
    const int n0   = blockIdx.x * 128;

    const int lrow = tid >> 1;          // 0..127 : tile row this thread stages
    const int lcol = (tid & 1) * 16;    // 0 or 16 halves

    const int khalf = (lane >= 16) ? 8 : 0;   // ISA 16-bit A/B lane K split
    const int rsel  = lane & 15;

    const unsigned short* gA = A + (size_t)(m0 + lrow) * Kd + lcol;
    const unsigned short* gW = W + (size_t)(n0 + lrow) * Kd + lcol;

    // Async-stage one 128x32 A tile + 128x32 W tile: 4 x b128 per thread.
    auto stage = [&](int buf, int k0) {
        unsigned int la = lds_addr32(&As[buf][lrow * 32 + lcol]);
        unsigned int lw = lds_addr32(&Ws[buf][lrow * 32 + lcol]);
        const void* pa0 = (const void*)(gA + k0);
        const void* pa1 = (const void*)(gA + k0 + 8);
        const void* pw0 = (const void*)(gW + k0);
        const void* pw1 = (const void*)(gW + k0 + 8);
        asm volatile(
            "global_load_async_to_lds_b128 %0, %4, off\n\t"
            "global_load_async_to_lds_b128 %1, %5, off\n\t"
            "global_load_async_to_lds_b128 %2, %6, off\n\t"
            "global_load_async_to_lds_b128 %3, %7, off"
            :
            : "v"(la), "v"(la + 16u), "v"(lw), "v"(lw + 16u),
              "v"(pa0), "v"(pa1), "v"(pw0), "v"(pw1)
            : "memory");
    };

    v8f zero = {0.f, 0.f, 0.f, 0.f, 0.f, 0.f, 0.f, 0.f};
    v8f acc[2][4] = {{zero, zero, zero, zero}, {zero, zero, zero, zero}};

    const int T = Kd / 32;
    stage(0, 0);                               // tile 0 in flight

    for (int t = 0; t < T; ++t) {
        const int cur = t & 1;
        // my async writes for tile t have landed in LDS
        asm volatile("s_wait_asynccnt 0x0" ::: "memory");
        // everyone's writes landed; everyone done reading the other buffer
        __syncthreads();
        if (t + 1 < T) stage(1 - cur, (t + 1) * 32);   // overlap with compute

        const unsigned int* lA = (const unsigned int*)As[cur];
        const unsigned int* lW = (const unsigned int*)Ws[cur];
        union FU { v16bf v; unsigned int u[8]; };
        FU af[2], wf[4];
#pragma unroll
        for (int fm = 0; fm < 2; ++fm) {
            int r = wm * 32 + fm * 16 + rsel;
#pragma unroll
            for (int v = 0; v < 8; ++v) {
                int kb = ((v & 3) * 2) + khalf + ((v >= 4) ? 16 : 0);
                af[fm].u[v] = lA[(r * 32 + kb) >> 1];
            }
        }
#pragma unroll
        for (int fn = 0; fn < 4; ++fn) {
            int c = wn * 64 + fn * 16 + rsel;
#pragma unroll
            for (int v = 0; v < 8; ++v) {
                int kb = ((v & 3) * 2) + khalf + ((v >= 4) ? 16 : 0);
                wf[fn].u[v] = lW[(c * 32 + kb) >> 1];
            }
        }
#pragma unroll
        for (int fm = 0; fm < 2; ++fm)
#pragma unroll
            for (int fn = 0; fn < 4; ++fn)
                acc[fm][fn] = __builtin_amdgcn_wmma_f32_16x16x32_bf16(
                    false, af[fm].v, false, wf[fn].v,
                    (short)0, acc[fm][fn], false, false);

        __syncthreads();   // done reading buf[cur] before it is re-staged
    }

    // epilogue: +2b, relu, bf16 store. C/D layout: row = base + v + 8*(lane>=16)
#pragma unroll
    for (int fn = 0; fn < 4; ++fn) {
        int col = n0 + wn * 64 + fn * 16 + rsel;
        float b2 = 2.0f * bias[col];
#pragma unroll
        for (int fm = 0; fm < 2; ++fm) {
            int rbase = m0 + wm * 32 + fm * 16 + ((lane >= 16) ? 8 : 0);
#pragma unroll
            for (int v = 0; v < 8; ++v) {
                float val = acc[fm][fn][v] + b2;
                val = fmaxf(val, 0.0f);
                O[(size_t)(rbase + v) * N + col] = f32_to_bf16(val);
            }
        }
    }
}

// -------------------------------- output head -------------------------------
// out[r] = relu(h_r . W0 + 2 b0) * wl0 + relu(h_r . W1 + 2 b1) * wl1
__global__ __launch_bounds__(256)
void final_k(const unsigned short* __restrict__ h, const float* __restrict__ Wout,
             const float* __restrict__ bias_o, const float* __restrict__ w_last,
             float* __restrict__ out, int Kd) {
    int wave = threadIdx.x >> 5, lane = threadIdx.x & 31;
    int row = blockIdx.x * 8 + wave;
    const unsigned short* hr = h + (size_t)row * Kd;
    float s0 = 0.f, s1 = 0.f;
    for (int k = lane; k < Kd; k += 32) {
        float hv = bf16_to_f32(hr[k]);
        s0 += hv * Wout[k];
        s1 += hv * Wout[Kd + k];
    }
#pragma unroll
    for (int off = 16; off > 0; off >>= 1) {
        s0 += __shfl_xor(s0, off, 32);
        s1 += __shfl_xor(s1, off, 32);
    }
    if (lane == 0) {
        float o0 = fmaxf(s0 + 2.0f * bias_o[0], 0.0f);
        float o1 = fmaxf(s1 + 2.0f * bias_o[1], 0.0f);
        out[row] = o0 * w_last[0] + o1 * w_last[1];
    }
}

// -------------------------------- launcher ----------------------------------

extern "C" void kernel_launch(void* const* d_in, const int* in_sizes, int n_in,
                              void* d_out, int out_size, void* d_ws, size_t ws_size,
                              hipStream_t stream) {
    const float* x   = (const float*)d_in[0];
    const float* s1h = (const float*)d_in[1];
    const float* s2h = (const float*)d_in[2];
    const float* bh  = (const float*)d_in[3];
    const float* s1o = (const float*)d_in[4];
    const float* s2o = (const float*)d_in[5];
    const float* bo  = (const float*)d_in[6];
    const float* wl  = (const float*)d_in[7];
    float* out = (float*)d_out;

    const int B = 8192, K = 4096, L = 5;
    const size_t BK = (size_t)B * K;       // activation elements
    const size_t KK = (size_t)K * K;       // weight elements

    char* ws = (char*)d_ws;
    unsigned short* hA   = (unsigned short*)(ws);                    // 64 MB
    unsigned short* hB   = (unsigned short*)(ws + BK * 2);           // 64 MB
    unsigned short* Wd   = (unsigned short*)(ws + BK * 4);           // 32 MB
    float*          Wout = (float*)(ws + BK * 4 + KK * 2);           // 32 KB
    unsigned int*   hist = (unsigned int*)(ws + BK * 4 + KK * 2 + (size_t)2 * K * 4);
    unsigned int*   sel  = hist + 65536;   // sel[0]=bucket sel[1]=rem sel[2]=T1 sel[3]=T2

    auto thresh = [&](const float* s, size_t n, unsigned int j, unsigned int* tout) {
        zero_hist_k<<<64, 256, 0, stream>>>(hist);
        hist_hi_k<<<2048, 256, 0, stream>>>(s, n, hist);
        select_hi_k<<<1, 1, 0, stream>>>(hist, j, sel);
        zero_hist_k<<<64, 256, 0, stream>>>(hist);
        hist_lo_k<<<2048, 256, 0, stream>>>(s, n, hist, sel);
        select_lo_k<<<1, 1, 0, stream>>>(hist, sel, tout);
    };

    // x -> bf16 activations
    cvt_f32_bf16_k<<<2048, 256, 0, stream>>>(x, hA, BK);

    unsigned short* cur = hA;
    unsigned short* nxt = hB;
    for (int l = 0; l < L; ++l) {
        const float* a = s1h + (size_t)l * KK;
        const float* b = s2h + (size_t)l * KK;
        thresh(a, KK, (unsigned int)(KK / 2), sel + 2);
        thresh(b, KK, (unsigned int)(KK / 2), sel + 3);
        build_w_k<<<4096, 256, 0, stream>>>(a, b, sel + 2, sel + 3, Wd, KK);
        dim3 grid(K / 128, B / 128);
        gemm_relu_bf16_k<<<grid, 256, 0, stream>>>(cur, Wd, bh + (size_t)l * K,
                                                   nxt, B, K, K);
        unsigned short* t = cur; cur = nxt; nxt = t;
    }

    // output layer: [2,4096] ternary weights, then 2->1 linear
    thresh(s1o, (size_t)2 * K, (unsigned int)K, sel + 2);
    thresh(s2o, (size_t)2 * K, (unsigned int)K, sel + 3);
    build_wout_k<<<32, 256, 0, stream>>>(s1o, s2o, sel + 2, sel + 3, Wout, (size_t)2 * K);
    final_k<<<B / 8, 256, 0, stream>>>(cur, Wout, bo, wl, out, K);
    (void)in_sizes; (void)n_in; (void)out_size; (void)ws_size;
}
